// Model2_65103114273350
// MI455X (gfx1250) — matrix operand order repaired and checked by reading, not compiled, not run on hardware
//
#include <hip/hip_runtime.h>
#include <hip/hip_bf16.h>
#include <math.h>

// ---------------------------------------------------------------------------
// Fused conv(18->32,3x3,SAME) + bias + ReLU + global-avg-pool via WMMA bf16.
// LDS staged channel-minor (bf16) + custom K-permutation so B fragments load
// as ds_load_b128 pairs. Then tiny MLP head + hierarchical mask + softmax.
// ---------------------------------------------------------------------------

typedef __attribute__((ext_vector_type(16))) __bf16 v16bf;
typedef __attribute__((ext_vector_type(8)))  __bf16 v8bf;
typedef __attribute__((ext_vector_type(4)))  __bf16 v4bf;
typedef __attribute__((ext_vector_type(2)))  __bf16 v2bf;
typedef __attribute__((ext_vector_type(8)))  float  v8f;
typedef __attribute__((ext_vector_type(4)))  float  v4f;

#define HW    224
#define CIN   18
#define COUT  32
#define KTOT  162                 // 18 * 3 * 3
#define NCH   6                   // K padded to 192 = 6 chunks of 32
#define NPIX  (HW * HW)

#define CP    24                  // channel stride (bf16): 48 B, 16B-aligned x-stride
#define XL    226                 // 1 left pad + 224 + 1 right pad
#define ROWS  10                  // y0-1 .. y0+8
#define XSTR  CP                  // x stride   (elements)
#define RSTR2 (XL * CP)           // row stride (elements) = 5424 -> 10848 B (16B mult.)
#define LDSE2 (ROWS * RSTR2)      // 54240 elements = 108480 bytes

// ---- K permutation -------------------------------------------------------
// Hardware K in [0,192) = h*16 + i.  Logical mapping:
//   h in 0..8 : tap = h,   c = i          (clean: 16 contiguous channels)
//   h == 9    : tap = i/2, c = 16+(i&1)   (leftover pairs, taps 0..7)
//   h == 10   : i<2 -> tap = 8, c = 16+i ; else zero
//   h == 11   : zero
// Flat weight offset inside [18][3][3] is c*9 + tap.
__host__ __device__ constexpr int permIdx(int K) {   // -1 => zero element
    const int h = K >> 4, i = K & 15;
    if (h <= 8)  return i * 9 + h;
    if (h == 9)  return (16 + (i & 1)) * 9 + (i >> 1);
    if (h == 10) return (i < 2) ? (16 + i) * 9 + 8 : -1;
    return -1;
}
// LDS element delta of a clean half h (tap = h): rows/x offsets of the tap.
__host__ __device__ constexpr int halfDelta(int h) {
    return (h / 3) * RSTR2 + (h % 3) * XSTR;         // c0 = 0
}
// LDS element delta of leftover pair for tap j (c = 16).
__host__ __device__ constexpr int pairDelta(int j) {
    return (j / 3) * RSTR2 + (j % 3) * XSTR + 16;
}

// One block: 8 waves <-> 8 consecutive output rows of one image.
// LDS limits us to 2 blocks/WGP, so trade occupancy for registers (no spills).
__global__ __launch_bounds__(256, 1)
void conv_pool_kernel(const float* __restrict__ x,
                      const float* __restrict__ w,
                      const float* __restrict__ bias,
                      float* __restrict__ gsum)      // [64][32] accumulators
{
    extern __shared__ __bf16 lds[];                  // LDSE2 elements

    const int tid  = threadIdx.x;
    const int lane = tid & 31;
    const int wv   = tid >> 5;                       // wave = row offset 0..7
    const int b    = blockIdx.x / (HW / 8);
    const int y0   = (blockIdx.x % (HW / 8)) * 8;
    const int g    = lane >> 4;
    const int n    = lane & 15;

    const float* xb = x + (size_t)b * CIN * NPIX;

    // ---- Stage rows y0-1..y0+8 channel-minor: lds[r][xl][c], xl = x+1 -----
    if (tid < ROWS) {                                 // zero pad columns xl=0,225
        v8bf z8 = {}; v4bf z4 = {};
        const int b0 = tid * RSTR2;
        *(v8bf*)&lds[b0 + 0] = z8;  *(v8bf*)&lds[b0 + 8] = z8;  *(v4bf*)&lds[b0 + 16] = z4;
        const int b1 = b0 + 225 * XSTR;
        *(v8bf*)&lds[b1 + 0] = z8;  *(v8bf*)&lds[b1 + 8] = z8;  *(v4bf*)&lds[b1 + 16] = z4;
    }
    for (int r = 0; r < ROWS; ++r) {                  // uniform row predicate
        const int yy = y0 - 1 + r;
        const bool vrow = (yy >= 0) && (yy < HW);
        const float* rowBase = xb + (size_t)yy * HW;
        for (int e = tid; e < CIN * (HW / 4); e += 256) {
            const int c  = e / (HW / 4);              // coalesced in x4
            const int x4 = e - c * (HW / 4);
            v4f v = {};
            if (vrow) v = *(const v4f*)(rowBase + (size_t)c * NPIX + x4 * 4);
            const int base = r * RSTR2 + (x4 * 4 + 1) * XSTR + c;
            lds[base + 0 * XSTR] = (__bf16)v[0];
            lds[base + 1 * XSTR] = (__bf16)v[1];
            lds[base + 2 * XSTR] = (__bf16)v[2];
            lds[base + 3 * XSTR] = (__bf16)v[3];
        }
    }
    __syncthreads();

    // ---- A (weight) fragments with the same K permutation, hoisted --------
    // A 16x32 fragment: element i <-> K = kc*32 + 16g + i (contiguous order, same as B).
    v16bf afrag[NCH][2];
    #pragma unroll
    for (int kc = 0; kc < NCH; ++kc) {
        #pragma unroll
        for (int mt = 0; mt < 2; ++mt) {
            const int m = mt * 16 + n;
            v16bf a;
            #pragma unroll
            for (int i = 0; i < 16; ++i) {
                if (kc < 5) {                         // h <= 9: always valid
                    const int K0 = kc * 32 + i;
                    const int p0 = permIdx(K0), p1 = permIdx(K0 + 16);
                    a[i] = (__bf16)w[m * KTOT + (g ? p1 : p0)];
                } else if (i < 2) {                   // K=160,161: g==0 only
                    const __bf16 wv2 = (__bf16)w[m * KTOT + (16 + i) * 9 + 8];
                    a[i] = g ? (__bf16)0.0f : wv2;
                } else {
                    a[i] = (__bf16)0.0f;
                }
            }
            afrag[kc][mt] = a;
        }
    }

    float bia0[8], bia1[8];
    #pragma unroll
    for (int r = 0; r < 8; ++r) {
        bia0[r] = bias[g * 8 + r];
        bia1[r] = bias[16 + g * 8 + r];
    }

    v8f acc0 = {}; v8f acc1 = {};
    const int commonBase = wv * RSTR2 + n * XSTR;     // + t*16*XSTR per tile

    for (int t = 0; t < HW / 16; ++t) {
        const int commonE = commonBase + t * 16 * XSTR;
        v8f c0 = {}; v8f c1 = {};
        #pragma unroll
        for (int kc = 0; kc < NCH; ++kc) {
            v16bf bfrag;
            if (kc < 4) {                             // both halves clean
                const int e = commonE + (g ? halfDelta(2 * kc + 1)
                                           : halfDelta(2 * kc));
                const v8bf lo = *(const v8bf*)&lds[e];        // ds_load_b128
                const v8bf hi = *(const v8bf*)&lds[e + 8];    // ds_load_b128
                #pragma unroll
                for (int i = 0; i < 8; ++i) { bfrag[i] = lo[i]; bfrag[8 + i] = hi[i]; }
            } else if (kc == 4) {                     // g==0 clean tap8, g==1 pairs
                const int e8 = commonE + halfDelta(8);
                const v8bf lo = *(const v8bf*)&lds[e8];
                const v8bf hi = *(const v8bf*)&lds[e8 + 8];
                __bf16 pv[16];
                #pragma unroll
                for (int j = 0; j < 8; ++j) {         // taps 0..7, c=16,17
                    const v2bf pr = *(const v2bf*)&lds[commonE + pairDelta(j)];
                    pv[2 * j] = pr[0]; pv[2 * j + 1] = pr[1];
                }
                #pragma unroll
                for (int i = 0; i < 16; ++i)
                    bfrag[i] = g ? pv[i] : ((i < 8) ? lo[i] : hi[i - 8]);
            } else {                                  // kc==5: K=160,161 (g==0)
                const v2bf pr = *(const v2bf*)&lds[commonE + pairDelta(8)];
                #pragma unroll
                for (int i = 0; i < 16; ++i) bfrag[i] = (__bf16)0.0f;
                bfrag[0] = g ? (__bf16)0.0f : pr[0];
                bfrag[1] = g ? (__bf16)0.0f : pr[1];
            }
            c0 = __builtin_amdgcn_wmma_f32_16x16x32_bf16(
                     false, afrag[kc][0], false, bfrag, (short)0, c0, false, false);
            c1 = __builtin_amdgcn_wmma_f32_16x16x32_bf16(
                     false, afrag[kc][1], false, bfrag, (short)0, c1, false, false);
        }
        #pragma unroll
        for (int r = 0; r < 8; ++r) {                 // bias+ReLU per pixel, pool
            acc0[r] += fmaxf(c0[r] + bia0[r], 0.0f);
            acc1[r] += fmaxf(c1[r] + bia1[r], 0.0f);
        }
    }

    // ---- Reduce 16 pixel-lanes per half-wave group, then atomics ----------
    #pragma unroll
    for (int r = 0; r < 8; ++r) {
        float s0 = acc0[r], s1 = acc1[r];
        #pragma unroll
        for (int off = 1; off < 16; off <<= 1) {
            s0 += __shfl_xor(s0, off, 32);
            s1 += __shfl_xor(s1, off, 32);
        }
        acc0[r] = s0; acc1[r] = s1;
    }
    if (n == 0) {
        const int chBase = g * 8;
        #pragma unroll
        for (int r = 0; r < 8; ++r) {
            atomicAdd(&gsum[b * COUT + chBase + r],      acc0[r]);
            atomicAdd(&gsum[b * COUT + 16 + chBase + r], acc1[r]);
        }
    }
}

// ---------------------------------------------------------------------------
// Head: pooled/50176, concat model1_pred, fc1+ReLU, fc2, mask, softmax.
// ---------------------------------------------------------------------------
__global__ void head_kernel(const float* __restrict__ gsum,
                            const float* __restrict__ pred,
                            const float* __restrict__ fc1w,
                            const float* __restrict__ fc1b,
                            const float* __restrict__ fc2w,
                            const float* __restrict__ fc2b,
                            float* __restrict__ out)
{
    const int bidx = threadIdx.x;
    if (bidx >= 64) return;

    float f[34];
    const float inv = 1.0f / (float)NPIX;
    #pragma unroll
    for (int i = 0; i < 32; ++i) f[i] = gsum[bidx * 32 + i] * inv;
    const float p0 = pred[bidx * 2 + 0];
    const float p1 = pred[bidx * 2 + 1];
    f[32] = p0; f[33] = p1;

    float h[64];
    for (int j = 0; j < 64; ++j) {
        float s = fc1b[j];
        #pragma unroll
        for (int i = 0; i < 34; ++i) s += f[i] * fc1w[j * 34 + i];
        h[j] = fmaxf(s, 0.0f);
    }

    const int idx = (p0 >= p1) ? 0 : 1;
    float lg[9];
    float mx = -INFINITY;
    for (int j = 0; j < 9; ++j) {
        float s = fc2b[j];
        #pragma unroll
        for (int i = 0; i < 64; ++i) s += h[i] * fc2w[j * 64 + i];
        const bool valid = (idx == 0) ? (j < 4) : (j >= 4);
        s = valid ? s : -INFINITY;
        lg[j] = s;
        mx = fmaxf(mx, s);
    }
    float den = 0.0f;
    #pragma unroll
    for (int j = 0; j < 9; ++j) { lg[j] = __expf(lg[j] - mx); den += lg[j]; }
    const float rden = 1.0f / den;
    #pragma unroll
    for (int j = 0; j < 9; ++j) out[bidx * 9 + j] = lg[j] * rden;
}

extern "C" void kernel_launch(void* const* d_in, const int* in_sizes, int n_in,
                              void* d_out, int out_size, void* d_ws, size_t ws_size,
                              hipStream_t stream)
{
    const float* x     = (const float*)d_in[0];
    const float* pred  = (const float*)d_in[1];
    const float* convw = (const float*)d_in[2];
    const float* convb = (const float*)d_in[3];
    const float* fc1w  = (const float*)d_in[4];
    const float* fc1b  = (const float*)d_in[5];
    const float* fc2w  = (const float*)d_in[6];
    const float* fc2b  = (const float*)d_in[7];

    float* gsum = (float*)d_ws;
    hipMemsetAsync(gsum, 0, 64 * COUT * sizeof(float), stream);

    const size_t ldsBytes = (size_t)LDSE2 * sizeof(__bf16);  // 108,480 B
    conv_pool_kernel<<<dim3(64 * (HW / 8)), dim3(256), ldsBytes, stream>>>(
        x, convw, convb, gsum);
    head_kernel<<<dim3(1), dim3(64), 0, stream>>>(gsum, pred, fc1w, fc1b, fc2w,
                                                  fc2b, (float*)d_out);
}
